// CausalSelfAttention_59502476919191
// MI455X (gfx1250) — compile-verified, matrix operand order checked by reading, not run on hardware
//
#include <hip/hip_runtime.h>

// CDNA5 / gfx1250 causal self-attention forward.
// - all matmuls via v_wmma_f32_16x16x32_f16 (wave32, f32 accumulate)
// - double-buffered LDS pipelines in the projection GEMMs
// - attention stages K/V blocks into LDS with global_load_async_to_lds_b128
//   (ASYNCcnt) shared by all 8 waves, double-buffered.

typedef _Float16 v16h __attribute__((ext_vector_type(16)));
typedef _Float16 v8h  __attribute__((ext_vector_type(8)));
typedef float    v8f  __attribute__((ext_vector_type(8)));

#define DEVI static __device__ __forceinline__

static constexpr int B_  = 4;
static constexpr int T_  = 2048;
static constexpr int C_  = 1024;
static constexpr int H_  = 16;
static constexpr int HD_ = 64;
static constexpr int M_  = B_ * T_;   // 8192 total rows

static constexpr int LDA = 40;  // GEMM LDS strides (f16 units; 80B rows)
static constexpr int LDB = 40;
static constexpr int LDP = 40;  // attention P staging stride
static constexpr int LDK = 72;  // attention K block stride (144B = 36 banks)
static constexpr int LDV = 40;  // attention V block stride

DEVI v8f wmma_f16f32(v16h a, v16h b, v8f c) {
  // D = A(16x32 f16) x B(32x16 f16) + C(16x16 f32)
  return __builtin_amdgcn_wmma_f32_16x16x32_f16(false, a, false, b,
                                                (short)0, c, false, false);
}

// Load a 16x32 operand fragment from [16-major][K] storage (row stride `ld`
// in f16 units).  Lanes 0-15: major = lane, K {0..7,16..23}; lanes 16-31:
// major = lane-16, K {8..15,24..31}.  Works for A (major = M row) and
// B (major = N column, i.e. rows of B^T).
DEVI v16h load_frag16(const _Float16* p0, int ld, int lane) {
  const int r  = lane & 15;
  const int kb = (lane & 16) ? 8 : 0;
  const _Float16* p = p0 + r * ld + kb;
  v8h lo = *(const v8h*)(p);
  v8h hi = *(const v8h*)(p + 16);
  return __builtin_shufflevector(lo, hi, 0, 1, 2, 3, 4, 5, 6, 7,
                                 8, 9, 10, 11, 12, 13, 14, 15);
}

// Async DMA: 16 bytes global -> LDS, tracked by ASYNCcnt.
DEVI void async_load_b128(unsigned int lds_addr, const void* gaddr) {
  asm volatile("global_load_async_to_lds_b128 %0, %1, off"
               :: "v"(lds_addr),
                  "v"((unsigned long long)(uintptr_t)gaddr)
               : "memory");
}
DEVI void wait_async0() { asm volatile("s_wait_asynccnt 0" ::: "memory"); }
DEVI void wait_ds0()    { asm volatile("s_wait_dscnt 0" ::: "memory"); }
DEVI unsigned int lds_lo32(const void* p) {
  return (unsigned int)(uintptr_t)p;  // generic LDS addr: offset in [31:0]
}

// ---------------------------------------------------------------------------
// Kernel 1: QKV projection.  blockIdx.z in {0,1,2} selects Q/K/V.
// 128x128 block tile, 8 waves, wave tile 32x64, double-buffered LDS.
// Q,K written f16 [B,H,T,64]; V written transposed f16 [B,H,64,T].
// ---------------------------------------------------------------------------
__global__ __launch_bounds__(256, 1)
void qkv_gemm_kernel(const float* __restrict__ x,
                     const float* __restrict__ Wq, const float* __restrict__ bq,
                     const float* __restrict__ Wk, const float* __restrict__ bk,
                     const float* __restrict__ Wv, const float* __restrict__ bv,
                     _Float16* __restrict__ qh, _Float16* __restrict__ kh,
                     _Float16* __restrict__ vth) {
  __shared__ __align__(16) _Float16 As[2][128 * LDA];  // [m][k]
  __shared__ __align__(16) _Float16 Bs[2][128 * LDB];  // [n][k] (W^T)

  const int tid  = threadIdx.x;
  const int lane = tid & 31;
  const int wid  = tid >> 5;
  const int wm   = wid & 3;   // 4 waves along M
  const int wn   = wid >> 2;  // 2 waves along N

  const int n0    = blockIdx.x * 128;
  const int m0    = blockIdx.y * 128;
  const int which = blockIdx.z;

  const float* W    = (which == 0) ? Wq : (which == 1) ? Wk : Wv;
  const float* bias = (which == 0) ? bq : (which == 1) ? bk : bv;

  const int arow = tid >> 3;        // 0..31 (+32*i)
  const int acol = (tid & 7) * 4;
  const int bkr  = tid >> 5;        // 0..7 (+8*i)
  const int bnc  = (tid & 31) * 4;

  float4 areg[4], breg[4];

  auto gload = [&](int kk) {
#pragma unroll
    for (int i = 0; i < 4; ++i)
      areg[i] = *(const float4*)(x + (size_t)(m0 + arow + 32 * i) * C_ + kk + acol);
#pragma unroll
    for (int i = 0; i < 4; ++i)
      breg[i] = *(const float4*)(W + (size_t)(kk + bkr + 8 * i) * C_ + n0 + bnc);
  };
  auto lstore = [&](int buf) {
#pragma unroll
    for (int i = 0; i < 4; ++i) {
      _Float16* d = &As[buf][(arow + 32 * i) * LDA + acol];
      d[0] = (_Float16)areg[i].x; d[1] = (_Float16)areg[i].y;
      d[2] = (_Float16)areg[i].z; d[3] = (_Float16)areg[i].w;
    }
#pragma unroll
    for (int i = 0; i < 4; ++i) {
      const int kr = bkr + 8 * i;
      Bs[buf][(bnc + 0) * LDB + kr] = (_Float16)breg[i].x;
      Bs[buf][(bnc + 1) * LDB + kr] = (_Float16)breg[i].y;
      Bs[buf][(bnc + 2) * LDB + kr] = (_Float16)breg[i].z;
      Bs[buf][(bnc + 3) * LDB + kr] = (_Float16)breg[i].w;
    }
  };

  v8f acc[2][4] = {};

  gload(0);
  lstore(0);
  __syncthreads();

  const int NK = C_ / 32;
  for (int it = 0; it < NK; ++it) {
    const int buf  = it & 1;
    const bool more = (it + 1) < NK;
    if (more) gload((it + 1) * 32);  // hide global latency behind WMMAs

    v16h af[2], bf[4];
#pragma unroll
    for (int mi = 0; mi < 2; ++mi)
      af[mi] = load_frag16(&As[buf][(wm * 32 + mi * 16) * LDA], LDA, lane);
#pragma unroll
    for (int ni = 0; ni < 4; ++ni)
      bf[ni] = load_frag16(&Bs[buf][(wn * 64 + ni * 16) * LDB], LDB, lane);
#pragma unroll
    for (int mi = 0; mi < 2; ++mi)
#pragma unroll
      for (int ni = 0; ni < 4; ++ni)
        acc[mi][ni] = wmma_f16f32(af[mi], bf[ni], acc[mi][ni]);

    if (more) lstore(buf ^ 1);
    __syncthreads();
  }

  // Epilogue: bias + scatter to head layouts.
  const int half8 = (lane & 16) ? 8 : 0;
  const int cn    = lane & 15;
#pragma unroll
  for (int mi = 0; mi < 2; ++mi) {
#pragma unroll
    for (int ni = 0; ni < 4; ++ni) {
#pragma unroll
      for (int r = 0; r < 8; ++r) {
        const int m = m0 + wm * 32 + mi * 16 + r + half8;
        const int n = n0 + wn * 64 + ni * 16 + cn;
        const float v = acc[mi][ni][r] + bias[n];
        const int b = m >> 11, t = m & (T_ - 1);
        const int h = n >> 6,  d = n & (HD_ - 1);
        const int bh = b * H_ + h;
        if (which == 2) {
          vth[((size_t)bh * HD_ + d) * T_ + t] = (_Float16)v;  // V^T
        } else {
          _Float16* dst = (which == 0) ? qh : kh;
          dst[((size_t)bh * T_ + t) * HD_ + d] = (_Float16)v;
        }
      }
    }
  }
}

// ---------------------------------------------------------------------------
// Kernel 2: causal flash attention.  grid = (T/128, B*H), 8 waves/block.
// K/V blocks (32 keys) staged cooperatively into LDS via async DMA,
// double-buffered; each wave owns 16 query rows with online softmax.
// ---------------------------------------------------------------------------
__global__ __launch_bounds__(256, 1)
void attn_kernel(const _Float16* __restrict__ qh,
                 const _Float16* __restrict__ kh,
                 const _Float16* __restrict__ vth,
                 _Float16* __restrict__ yh) {
  __shared__ __align__(16) _Float16 Ks[2][32 * LDK];   // [time 32][hd 64]
  __shared__ __align__(16) _Float16 Vs[2][64 * LDV];   // [hd 64][time 32]
  __shared__ __align__(16) _Float16 Ps[8][16 * LDP];   // per-wave P staging

  const int tid  = threadIdx.x;
  const int lane = tid & 31;
  const int wave = tid >> 5;
  const int bh   = blockIdx.y;
  const int qb0  = blockIdx.x * 128;
  const int qb   = qb0 + wave * 16;
  const int qbu  = __builtin_amdgcn_readfirstlane(qb);  // scalar copy

  const int half8 = (lane & 16) ? 8 : 0;
  const int cn    = lane & 15;
  const float scale = 0.125f;  // 1/sqrt(64)

  // cooperative staging slots (one b128 per thread per matrix)
  const int krow = tid >> 3;        // 0..31
  const int kcol = (tid & 7) * 8;   // f16 units
  const int vrow = tid >> 2;        // 0..63
  const int vcol = (tid & 3) * 8;
  const _Float16* kbase = kh  + (size_t)bh * T_ * HD_;
  const _Float16* vbase = vth + (size_t)bh * HD_ * T_;

  auto stage = [&](int buf, int k0) {
    async_load_b128(lds_lo32(&Ks[buf][krow * LDK + kcol]),
                    kbase + (size_t)(k0 + krow) * HD_ + kcol);
    async_load_b128(lds_lo32(&Vs[buf][vrow * LDV + vcol]),
                    vbase + (size_t)vrow * T_ + k0 + vcol);
  };

  const _Float16* qptr = qh + ((size_t)bh * T_ + qb) * HD_;
  const v16h qf0 = load_frag16(qptr,      HD_, lane);  // hd 0..31
  const v16h qf1 = load_frag16(qptr + 32, HD_, lane);  // hd 32..63

  v8f o0 = {}, o1 = {}, o2 = {}, o3 = {};
  float mst[8], lst[8];
#pragma unroll
  for (int r = 0; r < 8; ++r) { mst[r] = -1e30f; lst[r] = 0.f; }

  _Float16* pbase = &Ps[wave][0];
  const int nkbmax = qb0 / 32 + 4;  // covers keys for rows qb0..qb0+127

  stage(0, 0);
  wait_async0();
  __syncthreads();

  for (int kb = 0; kb < nkbmax; ++kb) {
    const int buf = kb & 1;
    const int k0  = kb * 32;
    const bool more = (kb + 1) < nkbmax;
    if (more) stage(buf ^ 1, k0 + 32);  // DMA next block during compute

    if (k0 <= qbu + 15) {  // scalar causal cut: this wave still has keys here
      const _Float16* ksb = &Ks[buf][0];
      v16h kf00 = load_frag16(ksb,                LDK, lane);
      v16h kf01 = load_frag16(ksb + 32,           LDK, lane);
      v16h kf10 = load_frag16(ksb + 16 * LDK,      LDK, lane);
      v16h kf11 = load_frag16(ksb + 16 * LDK + 32, LDK, lane);

      v8f s0 = {}, s1 = {};
      s0 = wmma_f16f32(qf0, kf00, s0);
      s0 = wmma_f16f32(qf1, kf01, s0);
      s1 = wmma_f16f32(qf0, kf10, s1);
      s1 = wmma_f16f32(qf1, kf11, s1);

#pragma unroll
      for (int r = 0; r < 8; ++r) {
        const int qrow = qb + r + half8;
        float x0 = s0[r] * scale;
        float x1 = s1[r] * scale;
        if (k0 + cn > qrow)      x0 = -1e30f;
        if (k0 + 16 + cn > qrow) x1 = -1e30f;
        float rm = fmaxf(x0, x1);
#pragma unroll
        for (int off = 1; off < 16; off <<= 1)
          rm = fmaxf(rm, __shfl_xor(rm, off, 32));
        const float mnew  = fmaxf(mst[r], rm);
        const float alpha = __expf(mst[r] - mnew);
        const float p0 = __expf(x0 - mnew);
        const float p1 = __expf(x1 - mnew);
        float rs = p0 + p1;
#pragma unroll
        for (int off = 1; off < 16; off <<= 1)
          rs += __shfl_xor(rs, off, 32);
        lst[r] = lst[r] * alpha + rs;
        mst[r] = mnew;
        o0[r] *= alpha; o1[r] *= alpha; o2[r] *= alpha; o3[r] *= alpha;
        // C-layout -> A-fragment layout via per-wave LDS bounce
        _Float16* pr = pbase + (r + half8) * LDP;
        pr[cn]      = (_Float16)p0;
        pr[16 + cn] = (_Float16)p1;
      }
      wait_ds0();  // same-wave LDS store->load

      const v16h pf = load_frag16(pbase, LDP, lane);
      const _Float16* vsb = &Vs[buf][0];
      v16h vf0 = load_frag16(vsb + 0 * 16 * LDV, LDV, lane);
      v16h vf1 = load_frag16(vsb + 1 * 16 * LDV, LDV, lane);
      v16h vf2 = load_frag16(vsb + 2 * 16 * LDV, LDV, lane);
      v16h vf3 = load_frag16(vsb + 3 * 16 * LDV, LDV, lane);
      o0 = wmma_f16f32(pf, vf0, o0);
      o1 = wmma_f16f32(pf, vf1, o1);
      o2 = wmma_f16f32(pf, vf2, o2);
      o3 = wmma_f16f32(pf, vf3, o3);
    }

    if (more) wait_async0();
    __syncthreads();
  }

  const int b = bh >> 4, h = bh & (H_ - 1);
#pragma unroll
  for (int r = 0; r < 8; ++r) {
    const float inv = 1.0f / lst[r];
    const int row = qb + r + half8;
    _Float16* ybase = yh + (size_t)(b * T_ + row) * C_ + h * HD_;
    ybase[0 * 16 + cn] = (_Float16)(o0[r] * inv);
    ybase[1 * 16 + cn] = (_Float16)(o1[r] * inv);
    ybase[2 * 16 + cn] = (_Float16)(o2[r] * inv);
    ybase[3 * 16 + cn] = (_Float16)(o3[r] * inv);
  }
}

// ---------------------------------------------------------------------------
// Kernel 3: output projection  out = y @ Wo + bo  (f32 output), pipelined.
// ---------------------------------------------------------------------------
__global__ __launch_bounds__(256, 1)
void out_gemm_kernel(const _Float16* __restrict__ yh,
                     const float* __restrict__ Wo,
                     const float* __restrict__ bo,
                     float* __restrict__ out) {
  __shared__ __align__(16) _Float16 As[2][128 * LDA];
  __shared__ __align__(16) _Float16 Bs[2][128 * LDB];

  const int tid  = threadIdx.x;
  const int lane = tid & 31;
  const int wid  = tid >> 5;
  const int wm   = wid & 3;
  const int wn   = wid >> 2;
  const int n0 = blockIdx.x * 128;
  const int m0 = blockIdx.y * 128;

  const int arow = tid >> 2;        // 0..63 (+64*i)
  const int acol = (tid & 3) * 8;
  const int bkr  = tid >> 5;
  const int bnc  = (tid & 31) * 4;

  v8h    areg[2];
  float4 breg[4];

  auto gload = [&](int kk) {
#pragma unroll
    for (int i = 0; i < 2; ++i)
      areg[i] = *(const v8h*)(yh + (size_t)(m0 + arow + 64 * i) * C_ + kk + acol);
#pragma unroll
    for (int i = 0; i < 4; ++i)
      breg[i] = *(const float4*)(Wo + (size_t)(kk + bkr + 8 * i) * C_ + n0 + bnc);
  };
  auto lstore = [&](int buf) {
#pragma unroll
    for (int i = 0; i < 2; ++i)
      *(v8h*)&As[buf][(arow + 64 * i) * LDA + acol] = areg[i];
#pragma unroll
    for (int i = 0; i < 4; ++i) {
      const int kr = bkr + 8 * i;
      Bs[buf][(bnc + 0) * LDB + kr] = (_Float16)breg[i].x;
      Bs[buf][(bnc + 1) * LDB + kr] = (_Float16)breg[i].y;
      Bs[buf][(bnc + 2) * LDB + kr] = (_Float16)breg[i].z;
      Bs[buf][(bnc + 3) * LDB + kr] = (_Float16)breg[i].w;
    }
  };

  v8f acc[2][4] = {};

  gload(0);
  lstore(0);
  __syncthreads();

  const int NK = C_ / 32;
  for (int it = 0; it < NK; ++it) {
    const int buf  = it & 1;
    const bool more = (it + 1) < NK;
    if (more) gload((it + 1) * 32);

    v16h af[2], bf[4];
#pragma unroll
    for (int mi = 0; mi < 2; ++mi)
      af[mi] = load_frag16(&As[buf][(wm * 32 + mi * 16) * LDA], LDA, lane);
#pragma unroll
    for (int ni = 0; ni < 4; ++ni)
      bf[ni] = load_frag16(&Bs[buf][(wn * 64 + ni * 16) * LDB], LDB, lane);
#pragma unroll
    for (int mi = 0; mi < 2; ++mi)
#pragma unroll
      for (int ni = 0; ni < 4; ++ni)
        acc[mi][ni] = wmma_f16f32(af[mi], bf[ni], acc[mi][ni]);

    if (more) lstore(buf ^ 1);
    __syncthreads();
  }

  const int half8 = (lane & 16) ? 8 : 0;
  const int cn    = lane & 15;
#pragma unroll
  for (int mi = 0; mi < 2; ++mi)
#pragma unroll
    for (int ni = 0; ni < 4; ++ni)
#pragma unroll
      for (int r = 0; r < 8; ++r) {
        const int m = m0 + wm * 32 + mi * 16 + r + half8;
        const int n = n0 + wn * 64 + ni * 16 + cn;
        out[(size_t)m * C_ + n] = acc[mi][ni][r] + bo[n];
      }
}

// ---------------------------------------------------------------------------
extern "C" void kernel_launch(void* const* d_in, const int* in_sizes, int n_in,
                              void* d_out, int out_size, void* d_ws,
                              size_t ws_size, hipStream_t stream) {
  (void)in_sizes; (void)n_in; (void)out_size; (void)ws_size;
  const float* x  = (const float*)d_in[0];
  const float* Wq = (const float*)d_in[1];
  const float* bq = (const float*)d_in[2];
  const float* Wk = (const float*)d_in[3];
  const float* bk = (const float*)d_in[4];
  const float* Wv = (const float*)d_in[5];
  const float* bv = (const float*)d_in[6];
  const float* Wo = (const float*)d_in[7];
  const float* bo = (const float*)d_in[8];

  const size_t elems = (size_t)M_ * C_;  // 8M f16 each -> 64 MB workspace
  _Float16* qh  = (_Float16*)d_ws;
  _Float16* kh  = qh + elems;
  _Float16* vth = kh + elems;
  _Float16* yh  = vth + elems;
  float* out = (float*)d_out;

  qkv_gemm_kernel<<<dim3(C_ / 128, M_ / 128, 3), 256, 0, stream>>>(
      x, Wq, bq, Wk, bk, Wv, bv, qh, kh, vth);
  attn_kernel<<<dim3(T_ / 128, B_ * H_), 256, 0, stream>>>(qh, kh, vth, yh);
  out_gemm_kernel<<<dim3(C_ / 128, M_ / 128), 256, 0, stream>>>(yh, Wo, bo, out);
}
